// MultiQueryAttention_24592982737067
// MI455X (gfx1250) — matrix operand order repaired and checked by reading, not compiled, not run on hardware
//
#include <hip/hip_runtime.h>

// MQA forward for MI455X / gfx1250 (wave32, WMMA f16 -> f32 accumulate).
// v2.1: register-blocked WMMA GEMM (32x64 per wave), flash attention with
//       4 heads/block sharing K/V tiles staged in LDS by the Tensor Data Mover
//       (6-arg tensor_load_to_lds builtin for this toolchain).

typedef _Float16 f16;
typedef __attribute__((ext_vector_type(16))) _Float16 v16h;
typedef __attribute__((ext_vector_type(8)))  _Float16 v8h;
typedef __attribute__((ext_vector_type(8)))  float    v8f;
typedef __attribute__((ext_vector_type(4)))  unsigned int v4u;
typedef __attribute__((ext_vector_type(8)))  int v8i;
typedef __attribute__((ext_vector_type(4)))  int v4i;
typedef unsigned int u32;

#define B_  2
#define S_  2048
#define H_  2048
#define NH_ 16
#define HD_ 128
#define M_  (B_*S_)

// ---- fragment loaders ------------------------------------------------------
// B-fragment (16-bit, 32x16): lane holds 16 contiguous f16 along K.
__device__ __forceinline__ v16h load_contig16(const f16* p) {
  v8h lo = *(const v8h*)p;
  v8h hi = *(const v8h*)(p + 8);
  v16h r;
#pragma unroll
  for (int i = 0; i < 8; ++i) { r[i] = lo[i]; r[i + 8] = hi[i]; }
  return r;
}

// A-fragment (16-bit, 16x32): per lane, elements 0..15 at p[0..15] (K = half*16..),
// contiguous along K.  p already includes half*16.
__device__ __forceinline__ v16h load_afrag(const f16* p) {
  v8h lo = *(const v8h*)p;
  v8h hi = *(const v8h*)(p + 8);
  v16h r;
#pragma unroll
  for (int i = 0; i < 8; ++i) { r[i] = lo[i]; r[i + 8] = hi[i]; }
  return r;
}

// ---- Tensor Data Mover: 2D tile load (6-arg builtin on this toolchain) -----
// Copies tile_d1 rows of tile_d0 f16 elements (row stride stride0 elements)
// from gptr into LDS at lds_off, contiguously.
__device__ __forceinline__ void tdm_load_2d(u32 lds_off, const f16* gptr,
                                            u32 tile_d0, u32 tile_d1,
                                            u32 tensor_d0, u32 tensor_d1,
                                            u32 stride0) {
  unsigned long long ga = (unsigned long long)(size_t)gptr;
  v4u g0;
  g0[0] = 1u;                                            // count=1, user D#
  g0[1] = lds_off;                                       // lds_addr (bytes)
  g0[2] = (u32)(ga & 0xffffffffu);                       // global_addr[31:0]
  g0[3] = (u32)((ga >> 32) & 0x01ffffffu) | (2u << 30);  // addr[56:32], type=2
  v8i g1;
  g1[0] = (int)(1u << 16);                               // mask=0, data_size=2B
  g1[1] = (int)((tensor_d0 & 0xffffu) << 16);            // tensor_dim0[15:0]
  g1[2] = (int)((tensor_d0 >> 16) | ((tensor_d1 & 0xffffu) << 16));
  g1[3] = (int)((tensor_d1 >> 16) | (tile_d0 << 16));    // tile_dim0
  g1[4] = (int)(tile_d1 & 0xffffu);                      // tile_dim1, tile_dim2=0
  g1[5] = (int)stride0;                                  // tensor_dim0_stride lo
  g1[6] = 0;                                             // stride hi, dim1_stride lo
  g1[7] = 0;
  v4i z4 = {0, 0, 0, 0};
  v8i z8 = {0, 0, 0, 0, 0, 0, 0, 0};
  __builtin_amdgcn_tensor_load_to_lds(g0, g1, z4, z4, z8, 0);
}

// ---- elementwise casts -----------------------------------------------------
__global__ void cast_f32_f16(const float* __restrict__ in, f16* __restrict__ out, int n) {
  int i = blockIdx.x * blockDim.x + threadIdx.x;
  if (i < n) out[i] = (f16)in[i];
}

// out[n*K + k] = (f16) in[k*N + n]
__global__ void transpose_cast(const float* __restrict__ in, f16* __restrict__ out,
                               int K, int N) {
  int n = blockIdx.x * 16 + threadIdx.x;
  int k = blockIdx.y * 16 + threadIdx.y;
  if (n < N && k < K) out[(size_t)n * K + k] = (f16)in[(size_t)k * N + n];
}

// ---- WMMA GEMM: C[M,N] = A[M,K] x Bt[N,K], 32x64 tile per wave -------------
// mode 0: f16 row-major   mode 1: f16 -> out[b][n][s] (V^T)   mode 2: f32
__global__ __launch_bounds__(128) void gemm16(const f16* __restrict__ A,
                                              const f16* __restrict__ Bt,
                                              f16* __restrict__ o16,
                                              float* __restrict__ o32,
                                              int M, int N, int K, int mode) {
  const int lane = threadIdx.x & 31;
  const int w    = threadIdx.x >> 5;
  const int hlf  = lane >> 4;
  const int l15  = lane & 15;
  const int rowBase = blockIdx.y * 64  + (w >> 1) * 32;   // wave: 32 rows
  const int colBase = blockIdx.x * 128 + (w & 1) * 64;    // wave: 64 cols

  const f16* ap = A  + (size_t)(rowBase + l15) * K + hlf * 16;
  const f16* bp = Bt + (size_t)(colBase + l15) * K + hlf * 16;
  const size_t row16 = (size_t)16 * K;

  v8f acc[2][4];
  v8f vzero = {};
#pragma unroll
  for (int i = 0; i < 2; ++i)
#pragma unroll
    for (int j = 0; j < 4; ++j) acc[i][j] = vzero;

  for (int k0 = 0; k0 < K; k0 += 32) {
    __builtin_prefetch(ap + k0 + 128, 0, 0);             // global_prefetch_b8
    __builtin_prefetch(bp + k0 + 128, 0, 0);
    v16h a0 = load_afrag(ap + k0);
    v16h a1 = load_afrag(ap + row16 + k0);
    v16h b[4];
#pragma unroll
    for (int j = 0; j < 4; ++j) b[j] = load_contig16(bp + (size_t)j * row16 + k0);
#pragma unroll
    for (int j = 0; j < 4; ++j) {
      acc[0][j] = __builtin_amdgcn_wmma_f32_16x16x32_f16(false, a0, false, b[j],
                                                         (short)0, acc[0][j], false, false);
      acc[1][j] = __builtin_amdgcn_wmma_f32_16x16x32_f16(false, a1, false, b[j],
                                                         (short)0, acc[1][j], false, false);
    }
  }

#pragma unroll
  for (int i = 0; i < 2; ++i)
#pragma unroll
    for (int j = 0; j < 4; ++j)
#pragma unroll
      for (int r = 0; r < 8; ++r) {
        int m = rowBase + i * 16 + r + hlf * 8;   // f32 C/D: row = vgpr + 8*half
        int n = colBase + j * 16 + l15;           // col = lane & 15
        float v = acc[i][j][r];
        if (mode == 0) {
          o16[(size_t)m * N + n] = (f16)v;
        } else if (mode == 1) {
          int bb = m / S_;
          int s  = m - bb * S_;
          o16[(size_t)bb * HD_ * S_ + (size_t)n * S_ + s] = (f16)v;
        } else {
          o32[(size_t)m * N + n] = v;
        }
      }
}

// ---- fused MQA flash attention --------------------------------------------
// Block = 4 waves = 4 heads sharing one 16-row query tile and one K/V tile.
// K/V tiles staged into LDS by the Tensor Data Mover (wave 0), published via
// s_wait_tensorcnt + workgroup barrier.  MQA: K/V shared across all heads.
__global__ __launch_bounds__(128) void mqa_flash(const f16* __restrict__ Q,
                                                 const f16* __restrict__ Kc,
                                                 const f16* __restrict__ Vt,
                                                 f16* __restrict__ O,
                                                 float scale) {
  __shared__ f16 ldsK[32 * 128];      // K tile: 32 keys x 128 d   (8 KB)
  __shared__ f16 ldsV[128 * 32];      // V^T tile: 128 d x 32 keys (8 KB)
  __shared__ f16 ldsP[4 * 16 * 32];   // per-wave P tiles          (4 KB)

  const int lane = threadIdx.x & 31;
  const int w    = threadIdx.x >> 5;
  const int hlf  = lane >> 4;
  const int l15  = lane & 15;
  const int qt   = blockIdx.x;
  const int head = blockIdx.y * 4 + w;
  const int b    = blockIdx.z;
  const size_t qrow0 = (size_t)b * S_ + qt * 16;

  // Per-wave Q fragments (whole HD=128) stay in registers.
  const f16* qb = Q + (qrow0 + l15) * H_ + (size_t)head * HD_;
  v16h qf[4];
#pragma unroll
  for (int dc = 0; dc < 4; ++dc) qf[dc] = load_afrag(qb + dc * 32 + hlf * 16);

  const f16* Kb = Kc + (size_t)b * S_ * HD_;
  const f16* Vb = Vt + (size_t)b * HD_ * S_;
  f16* myP = &ldsP[w * 16 * 32];
  const u32 ldsK_off = (u32)(size_t)(void*)ldsK;
  const u32 ldsV_off = (u32)(size_t)(void*)ldsV;

  float rmax[8], rsum[8];
  v8f ao[8];
  v8f vzero = {};
#pragma unroll
  for (int r = 0; r < 8; ++r) { rmax[r] = -__builtin_inff(); rsum[r] = 0.0f; }
#pragma unroll
  for (int c = 0; c < 8; ++c) ao[c] = vzero;

  for (int kt = 0; kt < S_; kt += 32) {
    if (w == 0) {
      // K tile: 32 rows (keys) x 128 f16, row stride HD -> LDS [32][128]
      tdm_load_2d(ldsK_off, Kb + (size_t)kt * HD_, HD_, 32, HD_, S_, HD_);
      // V^T tile: 128 rows (d) x 32 f16 (keys), row stride S -> LDS [128][32]
      tdm_load_2d(ldsV_off, Vb + kt, 32, HD_, S_, HD_, S_);
      __builtin_amdgcn_s_wait_tensorcnt(0);
    }
    __syncthreads();

#pragma unroll
    for (int sub = 0; sub < 2; ++sub) {
      v8f sacc = {};
#pragma unroll
      for (int dc = 0; dc < 4; ++dc) {
        // B[d][key] from LDS K tile: lane col = local key, contiguous 16 in d.
        v16h kf = load_contig16(&ldsK[(sub * 16 + l15) * 128 + dc * 32 + hlf * 16]);
        sacc = __builtin_amdgcn_wmma_f32_16x16x32_f16(false, qf[dc], false, kf,
                                                      (short)0, sacc, false, false);
      }
      // Online softmax; row m = r + 8*hlf lives across the 16 lanes of a half.
#pragma unroll
      for (int r = 0; r < 8; ++r) {
        float sv = sacc[r] * scale;
        float mx = sv;
        mx = fmaxf(mx, __shfl_xor(mx, 1, 32));
        mx = fmaxf(mx, __shfl_xor(mx, 2, 32));
        mx = fmaxf(mx, __shfl_xor(mx, 4, 32));
        mx = fmaxf(mx, __shfl_xor(mx, 8, 32));
        float nm = fmaxf(rmax[r], mx);
        float cr = __expf(rmax[r] - nm);
        float p  = __expf(sv - nm);
        float ps = p;
        ps += __shfl_xor(ps, 1, 32);
        ps += __shfl_xor(ps, 2, 32);
        ps += __shfl_xor(ps, 4, 32);
        ps += __shfl_xor(ps, 8, 32);
        rsum[r] = rsum[r] * cr + ps;
        rmax[r] = nm;
#pragma unroll
        for (int c = 0; c < 8; ++c) ao[c][r] = ao[c][r] * cr;
        myP[(r + hlf * 8) * 32 + sub * 16 + l15] = (f16)p;
      }
    }

    // P(16x32) as A-fragment (in-wave LDS RAW is in-order); V from LDS tile.
    v16h pf = load_afrag(&myP[l15 * 32 + hlf * 16]);
#pragma unroll
    for (int c = 0; c < 8; ++c) {
      v16h vf = load_contig16(&ldsV[(c * 16 + l15) * 32 + hlf * 16]);
      ao[c] = __builtin_amdgcn_wmma_f32_16x16x32_f16(false, pf, false, vf,
                                                     (short)0, ao[c], false, false);
    }
    __syncthreads();   // protect LDS K/V before next TDM overwrite
  }

#pragma unroll
  for (int r = 0; r < 8; ++r) {
    float inv = 1.0f / rsum[r];
    size_t m = qrow0 + r + hlf * 8;
#pragma unroll
    for (int c = 0; c < 8; ++c)
      O[m * H_ + (size_t)head * HD_ + c * 16 + l15] = (f16)(ao[c][r] * inv);
  }
}

// ---- launcher --------------------------------------------------------------
extern "C" void kernel_launch(void* const* d_in, const int* in_sizes, int n_in,
                              void* d_out, int out_size, void* d_ws, size_t ws_size,
                              hipStream_t stream) {
  (void)in_sizes; (void)n_in; (void)out_size; (void)ws_size;
  const float* x  = (const float*)d_in[0];
  const float* Wq = (const float*)d_in[1];
  const float* Wk = (const float*)d_in[2];
  const float* Wv = (const float*)d_in[3];
  const float* Wo = (const float*)d_in[4];
  float* out = (float*)d_out;

  char* ws = (char*)d_ws;
  size_t off = 0;
  auto alloc = [&](size_t elems) -> f16* {
    f16* p = (f16*)(ws + off);
    off += elems * sizeof(f16);
    off = (off + 255) & ~(size_t)255;
    return p;
  };
  f16* x16  = alloc((size_t)M_ * H_);    // x in f16
  f16* q16  = alloc((size_t)M_ * H_);    // Q [B*S, H]
  f16* k16  = alloc((size_t)M_ * HD_);   // K [B, S, HD]
  f16* vt16 = alloc((size_t)M_ * HD_);   // V^T [B, HD, S]
  f16* o16  = alloc((size_t)M_ * H_);    // attention output (pre-Wo)
  f16* wqt  = alloc((size_t)H_ * H_);    // Wq^T [N,K]
  f16* wkt  = alloc((size_t)H_ * HD_);
  f16* wvt  = alloc((size_t)H_ * HD_);
  f16* wot  = alloc((size_t)H_ * H_);

  const int n = M_ * H_;
  cast_f32_f16<<<dim3(n / 256), dim3(256), 0, stream>>>(x, x16, n);
  transpose_cast<<<dim3(H_ / 16, H_ / 16), dim3(16, 16), 0, stream>>>(Wq, wqt, H_, H_);
  transpose_cast<<<dim3(HD_ / 16, H_ / 16), dim3(16, 16), 0, stream>>>(Wk, wkt, H_, HD_);
  transpose_cast<<<dim3(HD_ / 16, H_ / 16), dim3(16, 16), 0, stream>>>(Wv, wvt, H_, HD_);
  transpose_cast<<<dim3(H_ / 16, H_ / 16), dim3(16, 16), 0, stream>>>(Wo, wot, H_, H_);

  // Projections (WMMA GEMMs): block = 4 waves, 64x128 tile.
  gemm16<<<dim3(H_ / 128, M_ / 64), 128, 0, stream>>>(x16, wqt, q16,  nullptr, M_, H_,  H_, 0);
  gemm16<<<dim3(HD_ / 128, M_ / 64), 128, 0, stream>>>(x16, wkt, k16,  nullptr, M_, HD_, H_, 0);
  gemm16<<<dim3(HD_ / 128, M_ / 64), 128, 0, stream>>>(x16, wvt, vt16, nullptr, M_, HD_, H_, 1);

  // Fused MQA flash attention: 4 heads/block share TDM-staged K/V tiles.
  mqa_flash<<<dim3(S_ / 16, NH_ / 4, B_), 128, 0, stream>>>(q16, k16, vt16, o16,
                                                            0.08838834764831845f);

  // Output projection -> f32 d_out.
  gemm16<<<dim3(H_ / 128, M_ / 64), 128, 0, stream>>>(o16, wot, nullptr, out, M_, H_, H_, 2);
}
